// BiMamba_29618094474256
// MI455X (gfx1250) — matrix operand order repaired and checked by reading, not compiled, not run on hardware
//
#include <hip/hip_runtime.h>
#include <hip/hip_bf16.h>

// ---------------- problem constants (match reference) ----------------
#define D_MODEL  768
#define D_INNER  1536
#define D_STATE  16
#define D_CONV   4
#define DT_RANK  48
#define NPROJ    (DT_RANK + 2 * D_STATE)   // 80
#define KDT_PAD  64                        // dt GEMM K padded 48 -> 64
#define BATCH    2
#define SEQ      2048
#define NT       (BATCH * SEQ)             // 4096 tokens

typedef __attribute__((ext_vector_type(16))) __bf16 v16bf;
typedef __attribute__((ext_vector_type(8)))  __bf16 v8bf;
typedef __attribute__((ext_vector_type(8)))  float  v8f;

union frag16 { v16bf v; v8bf h[2]; };

// ---------------- helpers ----------------
__device__ __forceinline__ __bf16 f2bf(float f) {
    // round-to-nearest-even fp32 -> bf16
    unsigned u = __builtin_bit_cast(unsigned, f);
    unsigned r = u + 0x7FFFu + ((u >> 16) & 1u);
    unsigned short s = (unsigned short)(r >> 16);
    return __builtin_bit_cast(__bf16, s);
}
__device__ __forceinline__ float siluf_(float x)     { return x / (1.0f + __expf(-x)); }
__device__ __forceinline__ float softplusf_(float x) { return (x > 20.0f) ? x : log1pf(__expf(x)); }

// ---------------- fp32 -> bf16 conversion kernels ----------------
__global__ void __launch_bounds__(256)
cvt_bf16_kernel(const float* __restrict__ src, __bf16* __restrict__ dst, int n)
{
    int i = blockIdx.x * blockDim.x + threadIdx.x;
    if (i < n) dst[i] = f2bf(src[i]);
}

// rows x Kp output; columns >= K zero-filled (zero-pads K to a multiple of 32)
__global__ void __launch_bounds__(256)
cvt_pad_bf16_kernel(const float* __restrict__ src, int ld_src, int K,
                    __bf16* __restrict__ dst, int Kp, int rows)
{
    int i = blockIdx.x * blockDim.x + threadIdx.x;
    if (i >= rows * Kp) return;
    int r = i / Kp, k = i - r * Kp;
    dst[i] = (k < K) ? f2bf(src[(size_t)r * ld_src + k]) : (__bf16)0.0f;
}

// ---------------- bf16 WMMA GEMM:  C[M,N] = act(A[M,K] @ W[N,K]^T + bias) ----------------
// Requirements: K % 32 == 0; all loads branch-free (row/col clamped to valid
// addresses; only the store is guarded). Wave tile 32(M) x 64(N): 2 A-frags x
// 4 B-frags -> 8 v_wmma per k-step. Block = 4 waves -> 128 x 64 tile.
__global__ void __launch_bounds__(128)
gemm_bf16_wmma(const __bf16* __restrict__ A, int lda,
               const __bf16* __restrict__ W, int ldw,
               float* __restrict__ C, int ldc,
               int M, int N, int K,
               const float* __restrict__ bias, int act)
{
    const int lane = threadIdx.x & 31;
    const int wave = threadIdx.x >> 5;
    const int hi   = lane >> 4;     // half-wave select
    const int lo   = lane & 15;
    const int m0   = blockIdx.y * 128 + wave * 32;
    const int n0   = blockIdx.x * 64;

    int ar0 = m0 + lo;      if (ar0 >= M) ar0 = M - 1;   // clamp: load-safe
    int ar1 = m0 + 16 + lo; if (ar1 >= M) ar1 = M - 1;
    const __bf16* Arow0 = A + (size_t)ar0 * lda + (hi << 3);
    const __bf16* Arow1 = A + (size_t)ar1 * lda + (hi << 3);
    const __bf16* Wcol[4];
#pragma unroll
    for (int nb = 0; nb < 4; ++nb) {
        int col = n0 + nb * 16 + lo; if (col >= N) col = N - 1;
        Wcol[nb] = W + (size_t)col * ldw + (hi << 3);
    }

    v8f acc[2][4] = {};

    for (int k0 = 0; k0 < K; k0 += 32) {
        // A fragment (16x32 MxK, ISA 7.12.2): lane holds row lane&15,
        // k = k0 + 8*hi + e  (e<8)  and  k = k0 + 16 + 8*hi + (e-8)
        frag16 a0, a1;
        a0.h[0] = *(const v8bf*)(Arow0 + k0);
        a0.h[1] = *(const v8bf*)(Arow0 + k0 + 16);
        a1.h[0] = *(const v8bf*)(Arow1 + k0);
        a1.h[1] = *(const v8bf*)(Arow1 + k0 + 16);
        __builtin_prefetch(Arow0 + k0 + 32, 0, 1);
        __builtin_prefetch(Arow1 + k0 + 32, 0, 1);
#pragma unroll
        for (int nb = 0; nb < 4; ++nb) {
            // B fragment (32x16 KxN): lane holds col lane&15, k = k0 + 8*hi + e (e<8) and k0 + 16 + 8*hi + (e-8)
            frag16 b;
            b.h[0] = *(const v8bf*)(Wcol[nb] + k0);
            b.h[1] = *(const v8bf*)(Wcol[nb] + k0 + 16);
            acc[0][nb] = __builtin_amdgcn_wmma_f32_16x16x32_bf16(
                false, a0.v, false, b.v, (short)0, acc[0][nb], false, false);
            acc[1][nb] = __builtin_amdgcn_wmma_f32_16x16x32_bf16(
                false, a1.v, false, b.v, (short)0, acc[1][nb], false, false);
        }
    }

    // D layout: VGPR v, lane L -> (m = mbase + v + 8*hi, n = lane&15)
#pragma unroll
    for (int mi = 0; mi < 2; ++mi) {
#pragma unroll
        for (int nb = 0; nb < 4; ++nb) {
            int n = n0 + nb * 16 + lo;
            if (n >= N) continue;
            float bv = bias ? bias[n] : 0.0f;
#pragma unroll
            for (int v = 0; v < 8; ++v) {
                int m = m0 + mi * 16 + v + (hi << 3);
                if (m < M) {
                    float r = acc[mi][nb][v] + bv;
                    if (act == 1) r = softplusf_(r);
                    C[(size_t)m * ldc + n] = r;
                }
            }
        }
    }
}

// ---------------- depthwise conv (causal fwd + anticausal bwd) + SiLU ----------------
// Writes both fp32 (for the scan) and bf16 (for the xproj GEMM) outputs.
__global__ void __launch_bounds__(256)
conv_silu_kernel(const float* __restrict__ xp,
                 const float* __restrict__ conv_w,
                 const float* __restrict__ conv_b,
                 float* __restrict__ xcf, float* __restrict__ xcb,
                 __bf16* __restrict__ xcfb, __bf16* __restrict__ xcbb)
{
    int idx = blockIdx.x * blockDim.x + threadIdx.x;
    if (idx >= NT * D_INNER) return;
    int c  = idx % D_INNER;
    int tg = idx / D_INNER;
    int t  = tg % SEQ;
    int b  = tg / SEQ;

    const float* base = xp + ((size_t)b * SEQ) * D_INNER + c;
    float sf = conv_b[c], sb = sf;
#pragma unroll
    for (int j = 0; j < D_CONV; ++j) {
        float w = conv_w[c * D_CONV + j];
        int tf = t - (D_CONV - 1) + j;
        if (tf >= 0)  sf = fmaf(w, base[(size_t)tf * D_INNER], sf);
        int tb = t + (D_CONV - 1) - j;
        if (tb < SEQ) sb = fmaf(w, base[(size_t)tb * D_INNER], sb);
    }
    sf = siluf_(sf);
    sb = siluf_(sb);
    xcf[idx] = sf;   xcfb[idx] = f2bf(sf);
    xcb[idx] = sb;   xcbb[idx] = f2bf(sb);
}

// ---------------- selective scan, fused skip+gate+average ----------------
__global__ void __launch_bounds__(256)
scan_kernel(const float* __restrict__ dt,    // [NT, D_INNER] post-softplus
            const float* __restrict__ proj,  // [NT, NPROJ]  (B at 48.., C at 64..)
            const float* __restrict__ xc,    // [NT, D_INNER]
            const float* __restrict__ A_log, // [D_INNER, D_STATE]
            const float* __restrict__ zg,    // [NT, D_INNER] gate pre-silu
            const float* __restrict__ Dp,    // [D_INNER]
            float* __restrict__ out,         // [NT, D_INNER]
            int dir)
{
    const int n  = threadIdx.x & 15;               // state index
    const int cl = threadIdx.x >> 4;               // local channel (0..15)
    const int c  = blockIdx.x * 16 + cl;
    const int b  = blockIdx.y;

    const float Aneg = -__expf(A_log[c * D_STATE + n]);
    const float Dc   = Dp[c];
    float h = 0.0f;

    for (int s = 0; s < SEQ; ++s) {
        const int t = dir ? (SEQ - 1 - s) : s;
        const size_t tg = (size_t)b * SEQ + t;
        const float d   = dt[tg * D_INNER + c];
        const float xv  = xc[tg * D_INNER + c];
        const float Bn  = proj[tg * NPROJ + DT_RANK + n];
        const float Cn  = proj[tg * NPROJ + DT_RANK + D_STATE + n];
        const float dA  = __expf(d * Aneg);
        h = fmaf(dA, h, d * xv * Bn);
        float yv = h * Cn;
#pragma unroll
        for (int off = 8; off > 0; off >>= 1)
            yv += __shfl_xor(yv, off, 16);
        if (n == 0) {
            const size_t oidx = tg * D_INNER + c;
            float g = 0.5f * siluf_(zg[oidx]) * (yv + Dc * xv);
            if (dir == 0) out[oidx] = g;
            else          out[oidx] += g;
        }
    }
}

// ---------------- launch ----------------
extern "C" void kernel_launch(void* const* d_in, const int* in_sizes, int n_in,
                              void* d_out, int out_size, void* d_ws, size_t ws_size,
                              hipStream_t stream) {
    (void)in_sizes; (void)n_in; (void)out_size; (void)ws_size;
    const float* x       = (const float*)d_in[0];
    const float* W_in    = (const float*)d_in[1];
    const float* conv_w  = (const float*)d_in[2];
    const float* conv_b  = (const float*)d_in[3];
    const float* W_xproj = (const float*)d_in[4];
    const float* W_dt    = (const float*)d_in[5];
    const float* b_dt    = (const float*)d_in[6];
    const float* A_log   = (const float*)d_in[7];
    const float* D_param = (const float*)d_in[8];
    const float* W_out   = (const float*)d_in[9];
    float* out = (float*)d_out;

    // ---- fp32 workspace ----
    float* ws = (float*)d_ws;
    size_t o = 0;
    float* xp    = ws + o; o += (size_t)NT * D_INNER;   // dead after conv -> reused as ycomb
    float* zg    = ws + o; o += (size_t)NT * D_INNER;
    float* xcf   = ws + o; o += (size_t)NT * D_INNER;
    float* xcb   = ws + o; o += (size_t)NT * D_INNER;
    float* dtf   = ws + o; o += (size_t)NT * D_INNER;
    float* dtb   = ws + o; o += (size_t)NT * D_INNER;
    float* projf = ws + o; o += (size_t)NT * NPROJ;
    float* projb = ws + o; o += (size_t)NT * NPROJ;
    float* ycomb = xp;   // alias: xp consumed by conv before scans write here

    // ---- bf16 workspace (carved after fp32 region; all offsets 16B-aligned) ----
    __bf16* bb = (__bf16*)(ws + o);
    size_t q = 0;
    __bf16* xbf    = bb + q; q += (size_t)NT * D_MODEL;
    __bf16* winbf  = bb + q; q += (size_t)2 * D_INNER * D_MODEL;
    __bf16* xcfb   = bb + q; q += (size_t)NT * D_INNER;
    __bf16* xcbb   = bb + q; q += (size_t)NT * D_INNER;
    __bf16* wxpbf  = bb + q; q += (size_t)NPROJ * D_INNER;
    __bf16* dtAf   = bb + q; q += (size_t)NT * KDT_PAD;
    __bf16* dtAb   = bb + q; q += (size_t)NT * KDT_PAD;
    __bf16* wdtbf  = bb + q; q += (size_t)D_INNER * KDT_PAD;
    __bf16* ycbf   = bb + q; q += (size_t)NT * D_INNER;
    __bf16* woutbf = bb + q; q += (size_t)D_MODEL * D_INNER;

    const dim3 gb(128, 1, 1);
#define CVT(src, dst, n) \
    cvt_bf16_kernel<<<((n) + 255) / 256, 256, 0, stream>>>((src), (dst), (n))

    // 0) one-time bf16 staging of GEMM operands
    CVT(x, xbf, NT * D_MODEL);
    CVT(W_in, winbf, 2 * D_INNER * D_MODEL);
    CVT(W_xproj, wxpbf, NPROJ * D_INNER);
    cvt_pad_bf16_kernel<<<(D_INNER * KDT_PAD + 255) / 256, 256, 0, stream>>>(
        W_dt, DT_RANK, DT_RANK, wdtbf, KDT_PAD, D_INNER);
    CVT(W_out, woutbf, D_MODEL * D_INNER);

    // 1) shared input projection: xp = x @ W_in[:1536].T ; z = x @ W_in[1536:].T
    gemm_bf16_wmma<<<dim3(D_INNER / 64, NT / 128), gb, 0, stream>>>(
        xbf, D_MODEL, winbf, D_MODEL, xp, D_INNER, NT, D_INNER, D_MODEL, nullptr, 0);
    gemm_bf16_wmma<<<dim3(D_INNER / 64, NT / 128), gb, 0, stream>>>(
        xbf, D_MODEL, winbf + (size_t)D_INNER * D_MODEL, D_MODEL, zg, D_INNER,
        NT, D_INNER, D_MODEL, nullptr, 0);

    // 2) depthwise conv + silu, both directions, fp32 + bf16 outputs
    conv_silu_kernel<<<(NT * D_INNER + 255) / 256, 256, 0, stream>>>(
        xp, conv_w, conv_b, xcf, xcb, xcfb, xcbb);

    // 3) x-projection -> (dt_raw | B | C), per direction
    gemm_bf16_wmma<<<dim3((NPROJ + 63) / 64, NT / 128), gb, 0, stream>>>(
        xcfb, D_INNER, wxpbf, D_INNER, projf, NPROJ, NT, NPROJ, D_INNER, nullptr, 0);
    gemm_bf16_wmma<<<dim3((NPROJ + 63) / 64, NT / 128), gb, 0, stream>>>(
        xcbb, D_INNER, wxpbf, D_INNER, projb, NPROJ, NT, NPROJ, D_INNER, nullptr, 0);

    // 4) dt = softplus(proj[:, :48] @ W_dt.T + b_dt), K zero-padded to 64
    cvt_pad_bf16_kernel<<<(NT * KDT_PAD + 255) / 256, 256, 0, stream>>>(
        projf, NPROJ, DT_RANK, dtAf, KDT_PAD, NT);
    cvt_pad_bf16_kernel<<<(NT * KDT_PAD + 255) / 256, 256, 0, stream>>>(
        projb, NPROJ, DT_RANK, dtAb, KDT_PAD, NT);
    gemm_bf16_wmma<<<dim3(D_INNER / 64, NT / 128), gb, 0, stream>>>(
        dtAf, KDT_PAD, wdtbf, KDT_PAD, dtf, D_INNER, NT, D_INNER, KDT_PAD, b_dt, 1);
    gemm_bf16_wmma<<<dim3(D_INNER / 64, NT / 128), gb, 0, stream>>>(
        dtAb, KDT_PAD, wdtbf, KDT_PAD, dtb, D_INNER, NT, D_INNER, KDT_PAD, b_dt, 1);

    // 5) selective scans; fused skip + silu(z) gate + 0.5 average into ycomb
    scan_kernel<<<dim3(D_INNER / 16, BATCH), 256, 0, stream>>>(
        dtf, projf, xcf, A_log, zg, D_param, ycomb, 0);
    scan_kernel<<<dim3(D_INNER / 16, BATCH), 256, 0, stream>>>(
        dtb, projb, xcb, A_log, zg, D_param, ycomb, 1);

    // 6) single output projection on the combined result
    CVT(ycomb, ycbf, NT * D_INNER);
    gemm_bf16_wmma<<<dim3(D_MODEL / 64, NT / 128), gb, 0, stream>>>(
        ycbf, D_INNER, woutbf, D_INNER, out, D_MODEL, NT, D_MODEL, D_INNER, nullptr, 0);
#undef CVT
}
